// LSTMCell__17557826306134
// MI455X (gfx1250) — matrix operand order repaired and checked by reading, not compiled, not run on hardware
//
#include <hip/hip_runtime.h>
#include <hip/hip_bf16.h>
#include <math.h>
#include <stdint.h>

// ---------------------------------------------------------------------------
// Peephole LSTM cell, B=4096, I=H=2048, on MI455X (gfx1250).
// Plan:
//   Phase 0: one-time fp32->bf16 conversion of x/h0/c0 and all weights,
//            packing the 4 gate weight matrices contiguously (gate-major).
//            (~200MB streamed once; all bf16 operands then fit in 192MB L2.)
//   Phase 1: bf16 WMMA GEMMs (f32 accum) into fused gate buffer G[4096][8192]
//            (gate order i,f,c,o), staging tiles via async global->LDS copies
//            (ASYNCcnt) with double-buffered LDS to overlap with v_wmma.
//   Phase 2: elementwise cell update, dependent c1 @ Wco^T GEMM, output gate.
// ---------------------------------------------------------------------------

typedef __attribute__((ext_vector_type(16))) __bf16 v16bf;
typedef __attribute__((ext_vector_type(8)))  __bf16 v8bf;
typedef __attribute__((ext_vector_type(4)))  __bf16 v4bf;
typedef __attribute__((ext_vector_type(8)))  float  v8f;
typedef int v4i_gcc __attribute__((vector_size(16)));   // matches builtin param

#define LSTM_B 4096
#define LSTM_I 2048
#define LSTM_H 2048
#define NTOT   8192   // 4 gates * H
#define KDIM   2048   // K is 2048 for every GEMM here

#define BM 128
#define BN 128
#define BK 64

#if defined(__has_builtin)
#if __has_builtin(__builtin_amdgcn_global_load_async_to_lds_b128)
#define HAVE_ASYNC_LDS 1
#endif
#endif

// 16-byte global -> LDS copy. Async (no VGPR round-trip, ASYNCcnt) when the
// toolchain exposes the gfx1250 builtin, else plain load+ds_store.
static __device__ __forceinline__ void cp16(const __bf16* g, __bf16* l) {
#ifdef HAVE_ASYNC_LDS
    __builtin_amdgcn_global_load_async_to_lds_b128(
        (__attribute__((address_space(1))) v4i_gcc*)(uintptr_t)g,
        (__attribute__((address_space(3))) v4i_gcc*)(uintptr_t)l,
        0, 0);
#else
    *(v8bf*)l = *(const v8bf*)g;
#endif
}

static __device__ __forceinline__ void async_wait_all() {
#ifdef HAVE_ASYNC_LDS
#if __has_builtin(__builtin_amdgcn_s_wait_asynccnt)
    __builtin_amdgcn_s_wait_asynccnt(0);
#else
    asm volatile("s_wait_asynccnt 0" ::: "memory");
#endif
#endif
}

static __device__ __forceinline__ float sigm(float x) {
    return 1.0f / (1.0f + __expf(-x));
}

// ---------------------------------------------------------------------------
// WMMA GEMM: C[m0.., n0..] (+)= A[M x K] * W^T  (+ bias)
// A bf16 row-major (M x K), W bf16 row-major (N x K): both K-major, matching
// v_wmma_f32_16x16x32_bf16's A (MxK) and B (KxN) lane layouts.
// ---------------------------------------------------------------------------
__global__ void __launch_bounds__(256)
lstm_gemm_wmma_bf16(const __bf16* __restrict__ A, const __bf16* __restrict__ W,
                    int K, float* __restrict__ C, int ldc,
                    const float* __restrict__ bias, int accum)
{
    alignas(16) __shared__ __bf16 As[2][BM * BK];   // 2 x 16KB
    alignas(16) __shared__ __bf16 Bs[2][BN * BK];   // 2 x 16KB

    const int tid = threadIdx.x;
    const int m0  = blockIdx.x * BM;
    const int n0  = blockIdx.y * BN;

    const int wave = tid >> 5;
    const int lane = tid & 31;
    const int wm   = wave & 3;          // 4 waves along M  (4*32 = 128)
    const int wn   = wave >> 2;         // 2 waves along N  (2*64 = 128)
    const int lr   = lane & 15;
    const int lhi  = lane >> 4;

    // Per-thread staging slots: tile = 128 rows x 64 cols bf16 = 1024 x 16B
    // chunks; 256 threads -> 4 chunks each (8 chunks per row).
    const __bf16* aSrc[4];
    const __bf16* bSrc[4];
    int           dOff[4];
    #pragma unroll
    for (int j = 0; j < 4; ++j) {
        int i   = tid + j * 256;
        int row = i >> 3;
        int col = (i & 7) << 3;
        aSrc[j] = A + (size_t)(m0 + row) * K + col;
        bSrc[j] = W + (size_t)(n0 + row) * K + col;
        dOff[j] = row * BK + col;
    }

    v8f acc[2][4] = {};

    const int nStages = K / BK;

    // Prime stage 0.
    #pragma unroll
    for (int j = 0; j < 4; ++j) {
        cp16(aSrc[j], &As[0][dOff[j]]);
        cp16(bSrc[j], &Bs[0][dOff[j]]);
    }

    for (int s = 0; s < nStages; ++s) {
        const int cur = s & 1;
        async_wait_all();
        __syncthreads();   // stage s visible to all; all done with stage s-1

        if (s + 1 < nStages) {
            const int nxt = cur ^ 1;
            const int kt  = (s + 1) * BK;
            #pragma unroll
            for (int j = 0; j < 4; ++j) {
                cp16(aSrc[j] + kt, &As[nxt][dOff[j]]);
                cp16(bSrc[j] + kt, &Bs[nxt][dOff[j]]);
            }
        }

        #pragma unroll
        for (int ks = 0; ks < BK; ks += 32) {
            v16bf afrag[2];
            v16bf bfrag[4];
            // A fragment: lanes 0-15 hold K[ks+0..7],K[ks+16..23] of row M=lr;
            //             lanes 16-31 hold K[ks+8..15],K[ks+24..31].
            #pragma unroll
            for (int m = 0; m < 2; ++m) {
                int r  = wm * 32 + m * 16 + lr;
                int k0 = ks + lhi * 8;
                v8bf lo = *(const v8bf*)&As[cur][r * BK + k0];
                v8bf hi = *(const v8bf*)&As[cur][r * BK + k0 + 16];
                afrag[m] = __builtin_shufflevector(lo, hi,
                    0,1,2,3,4,5,6,7,8,9,10,11,12,13,14,15);
            }
            // B fragment: lanes 0-15 hold K[ks+0..7],K[ks+16..23] of column N=lr;
            //             lanes 16-31 hold K[ks+8..15],K[ks+24..31].
            #pragma unroll
            for (int n = 0; n < 4; ++n) {
                int r  = wn * 64 + n * 16 + lr;   // W row == output column
                int k0 = ks + lhi * 8;
                v8bf b0 = *(const v8bf*)&Bs[cur][r * BK + k0];
                v8bf b1 = *(const v8bf*)&Bs[cur][r * BK + k0 + 16];
                bfrag[n] = __builtin_shufflevector(b0, b1,
                    0,1,2,3,4,5,6,7,8,9,10,11,12,13,14,15);
            }
            #pragma unroll
            for (int m = 0; m < 2; ++m)
                #pragma unroll
                for (int n = 0; n < 4; ++n)
                    acc[m][n] = __builtin_amdgcn_wmma_f32_16x16x32_bf16(
                        false, afrag[m], false, bfrag[n],
                        (short)0, acc[m][n], false, false);
        }
    }

    // Epilogue. C/D layout: VGPR e -> M = e (lanes 0-15) or e+8 (lanes 16-31),
    // N = lane%16.
    #pragma unroll
    for (int m = 0; m < 2; ++m) {
        #pragma unroll
        for (int n = 0; n < 4; ++n) {
            int col = n0 + wn * 64 + n * 16 + lr;
            #pragma unroll
            for (int e = 0; e < 8; ++e) {
                int row = m0 + wm * 32 + m * 16 + e + lhi * 8;
                size_t off = (size_t)row * ldc + col;
                float v = acc[m][n][e];
                if (accum) v += C[off];
                if (bias)  v += bias[col];
                C[off] = v;
            }
        }
    }
}

// ---------------------------------------------------------------------------
// One-time fp32 -> bf16 pack (8 elements / thread).
// ---------------------------------------------------------------------------
__global__ void __launch_bounds__(256)
f32_to_bf16(const float* __restrict__ src, __bf16* __restrict__ dst, int n8)
{
    int i = blockIdx.x * blockDim.x + threadIdx.x;
    if (i >= n8) return;
    size_t base = (size_t)i * 8;
    float4 a = *(const float4*)(src + base);
    float4 b = *(const float4*)(src + base + 4);
    v8bf o;
    o[0] = (__bf16)a.x; o[1] = (__bf16)a.y; o[2] = (__bf16)a.z; o[3] = (__bf16)a.w;
    o[4] = (__bf16)b.x; o[5] = (__bf16)b.y; o[6] = (__bf16)b.z; o[7] = (__bf16)b.w;
    *(v8bf*)(dst + base) = o;
}

// ---------------------------------------------------------------------------
// Pack per-gate bias sums: [bii+bhi | bif+bhf+bcf | bic+bhc | bio+bho+bco]
// ---------------------------------------------------------------------------
__global__ void __launch_bounds__(256)
lstm_bias_pack(const float* bii, const float* bhi,
               const float* bif, const float* bhf, const float* bcf,
               const float* bic, const float* bhc,
               const float* bio, const float* bho, const float* bco,
               float* biasTot)
{
    int n = blockIdx.x * blockDim.x + threadIdx.x;
    if (n >= NTOT) return;
    int g = n >> 11, j = n & 2047;
    float v;
    if      (g == 0) v = bii[j] + bhi[j];
    else if (g == 1) v = bif[j] + bhf[j] + bcf[j];
    else if (g == 2) v = bic[j] + bhc[j];
    else             v = bio[j] + bho[j] + bco[j];
    biasTot[n] = v;
}

// c1 = sig(Af)*c0 + sig(Ai)*tanh(Ac); emit c1 as f32 (output) and bf16 (GEMM A)
__global__ void __launch_bounds__(256)
lstm_elem_cell(const float* __restrict__ G, const float* __restrict__ c0,
               float* __restrict__ c1out, __bf16* __restrict__ c1b)
{
    int idx4 = blockIdx.x * blockDim.x + threadIdx.x;     // over B*H/4
    if (idx4 >= (LSTM_B * LSTM_H) / 4) return;
    int m  = idx4 >> 9;                                   // H/4 = 512
    int j4 = idx4 & 511;
    size_t base = (size_t)m * NTOT + j4 * 4;
    float4 ai  = *(const float4*)(G + base);
    float4 af  = *(const float4*)(G + base + 2048);
    float4 ac  = *(const float4*)(G + base + 4096);
    float4 c0v = *(const float4*)(c0 + (size_t)idx4 * 4);
    float4 r;
    r.x = sigm(af.x) * c0v.x + sigm(ai.x) * tanhf(ac.x);
    r.y = sigm(af.y) * c0v.y + sigm(ai.y) * tanhf(ac.y);
    r.z = sigm(af.z) * c0v.z + sigm(ai.z) * tanhf(ac.z);
    r.w = sigm(af.w) * c0v.w + sigm(ai.w) * tanhf(ac.w);
    *(float4*)(c1out + (size_t)idx4 * 4) = r;
    v4bf rb;
    rb[0] = (__bf16)r.x; rb[1] = (__bf16)r.y;
    rb[2] = (__bf16)r.z; rb[3] = (__bf16)r.w;
    *(v4bf*)(c1b + (size_t)idx4 * 4) = rb;
}

// o = sig(Ao); h1 = o*tanh(c1)
__global__ void __launch_bounds__(256)
lstm_elem_out(const float* __restrict__ G, const float* __restrict__ c1,
              float* __restrict__ og, float* __restrict__ h1)
{
    int idx4 = blockIdx.x * blockDim.x + threadIdx.x;
    if (idx4 >= (LSTM_B * LSTM_H) / 4) return;
    int m  = idx4 >> 9;
    int j4 = idx4 & 511;
    size_t base = (size_t)m * NTOT + j4 * 4;
    float4 ao  = *(const float4*)(G + base + 6144);
    float4 c1v = *(const float4*)(c1 + (size_t)idx4 * 4);
    float4 o, h;
    o.x = sigm(ao.x); o.y = sigm(ao.y); o.z = sigm(ao.z); o.w = sigm(ao.w);
    h.x = o.x * tanhf(c1v.x); h.y = o.y * tanhf(c1v.y);
    h.z = o.z * tanhf(c1v.z); h.w = o.w * tanhf(c1v.w);
    *(float4*)(og + (size_t)idx4 * 4) = o;
    *(float4*)(h1 + (size_t)idx4 * 4) = h;
}

extern "C" void kernel_launch(void* const* d_in, const int* in_sizes, int n_in,
                              void* d_out, int out_size, void* d_ws, size_t ws_size,
                              hipStream_t stream)
{
    const float* x   = (const float*)d_in[0];
    const float* h0  = (const float*)d_in[1];
    const float* c0  = (const float*)d_in[2];
    const float* Wii = (const float*)d_in[3];  const float* bii = (const float*)d_in[4];
    const float* Whi = (const float*)d_in[5];  const float* bhi = (const float*)d_in[6];
    const float* Wif = (const float*)d_in[7];  const float* bif = (const float*)d_in[8];
    const float* Whf = (const float*)d_in[9];  const float* bhf = (const float*)d_in[10];
    const float* Wcf = (const float*)d_in[11]; const float* bcf = (const float*)d_in[12];
    const float* Wic = (const float*)d_in[13]; const float* bic = (const float*)d_in[14];
    const float* Whc = (const float*)d_in[15]; const float* bhc = (const float*)d_in[16];
    const float* Wio = (const float*)d_in[17]; const float* bio = (const float*)d_in[18];
    const float* Who = (const float*)d_in[19]; const float* bho = (const float*)d_in[20];
    const float* Wco = (const float*)d_in[21]; const float* bco = (const float*)d_in[22];

    float* out_og = (float*)d_out;
    float* out_h1 = out_og + (size_t)LSTM_B * LSTM_H;
    float* out_c1 = out_og + 2 * (size_t)LSTM_B * LSTM_H;

    // ---- workspace layout -------------------------------------------------
    const size_t actN = (size_t)LSTM_B * LSTM_H;      // 8M elems
    const size_t wN   = (size_t)LSTM_H * KDIM;        // 4M elems per matrix
    char* p = (char*)d_ws;
    float*  G       = (float*)p;            p += (size_t)LSTM_B * NTOT * 4;  // 134MB
    float*  biasTot = (float*)p;            p += NTOT * 4;
    __bf16* xb      = (__bf16*)p;           p += actN * 2;
    __bf16* hb      = (__bf16*)p;           p += actN * 2;
    __bf16* cb      = (__bf16*)p;           p += actN * 2;
    __bf16* c1b     = (__bf16*)p;           p += actN * 2;
    __bf16* Wx      = (__bf16*)p;           p += 4 * wN * 2;   // [ii|if|ic|io]
    __bf16* Wh      = (__bf16*)p;           p += 4 * wN * 2;   // [hi|hf|hc|ho]
    __bf16* Wcfb    = (__bf16*)p;           p += wN * 2;
    __bf16* Wcob    = (__bf16*)p;           p += wN * 2;
    (void)ws_size;

    dim3 blk(256);
    dim3 gridFull(LSTM_B / BM, NTOT / BN);     // 32 x 64
    dim3 gridGate(LSTM_B / BM, LSTM_H / BN);   // 32 x 16
    const int actB = (int)(actN / 8 / 256);    // conversion blocks (activations)
    const int wB   = (int)(wN / 8 / 256);      // conversion blocks (weights)
    const int elemBlocks = (int)(actN / 4 / 256);

    // ---- phase 0: pack everything to bf16 --------------------------------
    f32_to_bf16<<<actB, blk, 0, stream>>>(x,  xb, (int)(actN / 8));
    f32_to_bf16<<<actB, blk, 0, stream>>>(h0, hb, (int)(actN / 8));
    f32_to_bf16<<<actB, blk, 0, stream>>>(c0, cb, (int)(actN / 8));
    f32_to_bf16<<<wB, blk, 0, stream>>>(Wii, Wx + 0 * wN, (int)(wN / 8));
    f32_to_bf16<<<wB, blk, 0, stream>>>(Wif, Wx + 1 * wN, (int)(wN / 8));
    f32_to_bf16<<<wB, blk, 0, stream>>>(Wic, Wx + 2 * wN, (int)(wN / 8));
    f32_to_bf16<<<wB, blk, 0, stream>>>(Wio, Wx + 3 * wN, (int)(wN / 8));
    f32_to_bf16<<<wB, blk, 0, stream>>>(Whi, Wh + 0 * wN, (int)(wN / 8));
    f32_to_bf16<<<wB, blk, 0, stream>>>(Whf, Wh + 1 * wN, (int)(wN / 8));
    f32_to_bf16<<<wB, blk, 0, stream>>>(Whc, Wh + 2 * wN, (int)(wN / 8));
    f32_to_bf16<<<wB, blk, 0, stream>>>(Who, Wh + 3 * wN, (int)(wN / 8));
    f32_to_bf16<<<wB, blk, 0, stream>>>(Wcf, Wcfb, (int)(wN / 8));
    f32_to_bf16<<<wB, blk, 0, stream>>>(Wco, Wcob, (int)(wN / 8));

    lstm_bias_pack<<<(NTOT + 255) / 256, blk, 0, stream>>>(
        bii, bhi, bif, bhf, bcf, bic, bhc, bio, bho, bco, biasTot);

    // ---- phase 1: gate pre-activations -----------------------------------
    // G = x @ [Wii|Wif|Wic|Wio]^T + biasTot
    lstm_gemm_wmma_bf16<<<gridFull, blk, 0, stream>>>(
        xb, Wx, KDIM, G, NTOT, biasTot, 0);
    // G += h0 @ [Whi|Whf|Whc|Who]^T
    lstm_gemm_wmma_bf16<<<gridFull, blk, 0, stream>>>(
        hb, Wh, KDIM, G, NTOT, nullptr, 1);
    // G[:, f-gate] += c0 @ Wcf^T
    lstm_gemm_wmma_bf16<<<gridGate, blk, 0, stream>>>(
        cb, Wcfb, KDIM, G + LSTM_H, NTOT, nullptr, 1);

    // ---- phase 2: cell state, dependent GEMM, outputs --------------------
    lstm_elem_cell<<<elemBlocks, blk, 0, stream>>>(G, c0, out_c1, c1b);

    lstm_gemm_wmma_bf16<<<gridGate, blk, 0, stream>>>(
        c1b, Wcob, KDIM, G + 3 * LSTM_H, NTOT, nullptr, 1);

    lstm_elem_out<<<elemBlocks, blk, 0, stream>>>(G, out_c1, out_og, out_h1);
}
